// NonLinearActorNet_24859270710028
// MI455X (gfx1250) — matrix operand order repaired and checked by reading, not compiled, not run on hardware
//
#include <hip/hip_runtime.h>
#include <stdint.h>

#define BATCH 2048
#define DIN   512
#define HID   512
#define NACT  18
#define NTASK 16
#define APITCH 520   // halves per row in LDS A / H tile (pad avoids bank conflicts)

typedef __attribute__((ext_vector_type(16))) __bf16 bf16x16;
typedef __attribute__((ext_vector_type(8)))  float  floatx8;

union FragAB { uint4 q[2]; bf16x16 v; };

static __device__ __forceinline__ unsigned short f32_bf16(float f) {
    unsigned u = __float_as_uint(f);
    u += 0x7FFFu + ((u >> 16) & 1u);           // round-to-nearest-even
    return (unsigned short)(u >> 16);
}
static __device__ __forceinline__ float bf16_f32(unsigned short h) {
    return __uint_as_float(((unsigned)h) << 16);
}

// ---------------------------------------------------------------- kernel 1
// Bin samples by task: perm[] holds sample indices grouped by task,
// bin_start[t]..bin_start[t+1] delimits task t's group.
__global__ void bin_tasks(const int* __restrict__ task,
                          int* __restrict__ bin_start,
                          int* __restrict__ perm) {
    __shared__ int cnt[NTASK], cur[NTASK], off[NTASK + 1];
    int tid = threadIdx.x;
    if (tid < NTASK) { cnt[tid] = 0; cur[tid] = 0; }
    __syncthreads();
    for (int b = tid; b < BATCH; b += blockDim.x) atomicAdd(&cnt[task[b]], 1);
    __syncthreads();
    if (tid == 0) {
        int a = 0;
        for (int t = 0; t < NTASK; ++t) { off[t] = a; a += cnt[t]; }
        off[NTASK] = a;
    }
    __syncthreads();
    for (int b = tid; b < BATCH; b += blockDim.x) {
        int t = task[b];
        int p = off[t] + atomicAdd(&cur[t], 1);
        perm[p] = b;
    }
    if (tid <= NTASK) bin_start[tid] = off[tid];
}

// ---------------------------------------------------------------- kernel 2
// Convert W1 (f32 [T][D][H]) into bf16 pre-swizzled WMMA B-fragment layout.
// Fragment unit: (t, ct, ks) -> 32 lanes x 8 dwords (16 bf16 per lane).
// Lane l: column h = ct*16 + (l&15); K chunk = ks*32 + (l<16 ? 0 : 16) + {0..15},
// packed as dword j = [K=2j (lo) | K=2j+1 (hi)].
__global__ void swizzle_w1(const float* __restrict__ W1,
                           unsigned int* __restrict__ frag) {
    int tid = blockIdx.x * blockDim.x + threadIdx.x;   // 16*32*16*32 = 262144
    int l  = tid & 31;
    int ks = (tid >> 5) & 15;
    int ct = (tid >> 9) & 31;
    int t  = tid >> 14;
    int d0 = ks * 32 + ((l & 16) ? 16 : 0);
    int h  = ct * 16 + (l & 15);
    const float* src = W1 + (size_t)t * (DIN * HID) + h;
    unsigned int o[8];
#pragma unroll
    for (int j = 0; j < 8; ++j) {
        unsigned short lo = f32_bf16(src[(size_t)(d0 + 2 * j)     * HID]);
        unsigned short hi = f32_bf16(src[(size_t)(d0 + 2 * j + 1) * HID]);
        o[j] = (unsigned)lo | ((unsigned)hi << 16);
    }
    uint4* dst = (uint4*)(frag + (size_t)tid * 8);
    dst[0] = make_uint4(o[0], o[1], o[2], o[3]);
    dst[1] = make_uint4(o[4], o[5], o[6], o[7]);
}

// ---------------------------------------------------------------- kernel 3
// Fused: per-task 16-row tile GEMM (bf16 WMMA) -> ReLU -> fc2 -> log-softmax.
__global__ __launch_bounds__(256) void actor_fused(
    const float* __restrict__ xs,  const float* __restrict__ b1,
    const float* __restrict__ W2,  const float* __restrict__ b2,
    const int*   __restrict__ action,
    const int*   __restrict__ bin_start, const int* __restrict__ perm,
    const unsigned int* __restrict__ w1frag,
    float* __restrict__ out) {

    __shared__ __align__(16) unsigned short sA[16 * APITCH];  // x-rows (bf16), reused for h
    __shared__ float sW2[HID * NACT];                          // W2[t]
    __shared__ int   sSid[16];

    // resolve (task, tile) for this block
    int g = blockIdx.x, t = -1, tile = 0, acc = 0;
    for (int tt = 0; tt < NTASK; ++tt) {
        int n  = bin_start[tt + 1] - bin_start[tt];
        int nt = (n + 15) >> 4;
        if (g < acc + nt) { t = tt; tile = g - acc; break; }
        acc += nt;
    }
    if (t < 0) return;

    int tid  = threadIdx.x;
    int b0   = bin_start[t] + tile * 16;
    int bend = bin_start[t + 1];
    if (tid < 16) {
        int pr = b0 + tid;
        sSid[tid] = (pr < bend) ? perm[pr] : -1;
    }
    __syncthreads();

    // stage A: 16 gathered x-rows, f32 -> bf16 (invalid rows zero)
    for (int idx = tid; idx < 16 * 128; idx += 256) {
        int row = idx >> 7, c4 = idx & 127;
        int sid = sSid[row];
        unsigned short h0 = 0, h1 = 0, h2 = 0, h3 = 0;
        if (sid >= 0) {
            float4 x = *(const float4*)(xs + (size_t)sid * DIN + c4 * 4);
            h0 = f32_bf16(x.x); h1 = f32_bf16(x.y);
            h2 = f32_bf16(x.z); h3 = f32_bf16(x.w);
        }
        unsigned short* p = sA + row * APITCH + c4 * 4;
        p[0] = h0; p[1] = h1; p[2] = h2; p[3] = h3;
    }
    // stage W2[t]
    {
        const float* src = W2 + (size_t)t * (HID * NACT);
        for (int idx = tid; idx < HID * NACT; idx += 256) sW2[idx] = src[idx];
    }
    __syncthreads();

    int wave = tid >> 5, lane = tid & 31;
    int half = (lane & 16) ? 1 : 0;
    int lm   = lane & 15;

    // fc1: wave owns columns [wave*64, wave*64+64) = 4 N-tiles
    floatx8 accv[4] = {floatx8{}, floatx8{}, floatx8{}, floatx8{}};
    const unsigned short* arow = sA + lm * APITCH;

    for (int ks = 0; ks < 16; ++ks) {
        FragAB fa;
        int kb = ks * 32 + half * 16;                 // A: K chunk kb..kb+15
        fa.q[0] = *(const uint4*)(arow + kb);
        fa.q[1] = *(const uint4*)(arow + kb + 8);
#pragma unroll
        for (int tt = 0; tt < 4; ++tt) {
            FragAB fb;
            const unsigned int* p =
                w1frag + ((size_t)((t * 32 + wave * 4 + tt) * 16 + ks) * 32 + lane) * 8;
            fb.q[0] = *(const uint4*)(p);
            fb.q[1] = *(const uint4*)(p + 4);
            accv[tt] = __builtin_amdgcn_wmma_f32_16x16x32_bf16(
                false, fa.v, false, fb.v, (short)0, accv[tt], false, false);
        }
    }

    // bias + ReLU into registers
    float hreg[4][8];
#pragma unroll
    for (int tt = 0; tt < 4; ++tt) {
        int col  = wave * 64 + tt * 16 + lm;
        float bv = b1[t * HID + col];
#pragma unroll
        for (int r = 0; r < 8; ++r) {
            float v = accv[tt][r] + bv;
            hreg[tt][r] = v > 0.f ? v : 0.f;
        }
    }
    __syncthreads();   // everyone done reading sA as A-input
#pragma unroll
    for (int tt = 0; tt < 4; ++tt) {
        int col = wave * 64 + tt * 16 + lm;
#pragma unroll
        for (int r = 0; r < 8; ++r) {
            int m = r + half * 8;                     // C layout: M = r + 8*(lane>=16)
            sA[m * APITCH + col] = f32_bf16(hreg[tt][r]);
        }
    }
    __syncthreads();

    // fc2 + log-softmax: wave handles rows 2w, 2w+1; lane a<18 owns one logit
    for (int r2 = 0; r2 < 2; ++r2) {
        int row = wave * 2 + r2;
        int sid = sSid[row];
        float dot = 0.f;
        if (lane < NACT) {
            const unsigned short* hrow = sA + row * APITCH;
#pragma unroll 4
            for (int k = 0; k < HID; ++k)
                dot = fmaf(bf16_f32(hrow[k]), sW2[k * NACT + lane], dot);
        }
        float logit = (lane < NACT) ? dot + b2[t * NACT + lane] : -INFINITY;
        float m = logit;
        for (int o = 16; o >= 1; o >>= 1) m = fmaxf(m, __shfl_xor(m, o));
        float e  = (lane < NACT) ? __expf(logit - m) : 0.f;
        float s  = e;
        float wl = (lane < NACT) ? e * logit : 0.f;
        for (int o = 16; o >= 1; o >>= 1) { s += __shfl_xor(s, o); wl += __shfl_xor(wl, o); }
        float logZ = m + __logf(s);
        float logp = logit - logZ;
        if (sid >= 0) {
            int   act = action[sid];
            float lpa = __shfl(logp, act);
            float ent = logZ - wl / s;                // -sum p*logp
            if (lane == 0) {
                out[sid]             = (float)act;
                out[BATCH + sid]     = lpa;
                out[2 * BATCH + sid] = ent;
            }
        }
    }
}

// ---------------------------------------------------------------- launch
extern "C" void kernel_launch(void* const* d_in, const int* in_sizes, int n_in,
                              void* d_out, int out_size, void* d_ws, size_t ws_size,
                              hipStream_t stream) {
    const float* xs  = (const float*)d_in[0];
    const float* W1  = (const float*)d_in[1];
    const float* b1  = (const float*)d_in[2];
    const float* W2  = (const float*)d_in[3];
    const float* b2  = (const float*)d_in[4];
    const int*   tsk = (const int*)d_in[5];
    const int*   act = (const int*)d_in[6];
    float* out = (float*)d_out;

    char* ws = (char*)d_ws;
    int* bin_start        = (int*)ws;                  // NTASK+1 = 17 ints
    int* perm             = (int*)(ws + 256);          // 2048 ints
    unsigned int* w1frag  = (unsigned int*)(ws + 65536); // 8 MB swizzled bf16 W1

    bin_tasks <<<1,    256, 0, stream>>>(tsk, bin_start, perm);
    swizzle_w1<<<1024, 256, 0, stream>>>(W1, w1frag);
    actor_fused<<<144, 256, 0, stream>>>(xs, b1, W2, b2, act,
                                         bin_start, perm, w1frag, out);
}
